// AsymmetricQuantLinear_51694226374740
// MI455X (gfx1250) — matrix operand order repaired and checked by reading, not compiled, not run on hardware
//
#include <hip/hip_runtime.h>

// ---- CDNA5 WMMA types -------------------------------------------------------
typedef __attribute__((ext_vector_type(16))) _Float16 v16h;  // A/B fragment (8 VGPRs)
typedef __attribute__((ext_vector_type(8)))  float    v8f;   // C/D fragment (8 VGPRs)
typedef __attribute__((ext_vector_type(8)))  _Float16 v8h;   // 16-byte LDS chunk
typedef __attribute__((ext_vector_type(2)))  _Float16 v2h;

// Problem constants (reference: M,K,N = 4096,4096,11008; PACKED_K = 2048)
#define MDIM 4096
#define KDIM 4096
#define NDIM 11008

// Tiling: 256x128 output tile per block (amortize weight dequant over 2x M),
// K-tile = 32, double-buffered LDS (48 KB) -> one barrier per K-tile.
#define BM 256
#define BN 128
#define BK 32

// ---------------------------------------------------------------------------
// Pre-pass: S[m] = sum_k x[m,k]  (one wave32 per row, 8 rows per block)
// ---------------------------------------------------------------------------
__global__ __launch_bounds__(256) void
rowsum_kernel(const float* __restrict__ x, float* __restrict__ S)
{
    const int row  = blockIdx.x * 8 + (threadIdx.x >> 5);
    const int lane = threadIdx.x & 31;
    const float* xr = x + (size_t)row * KDIM;

    float acc = 0.f;
    for (int i = lane * 4; i < KDIM; i += 32 * 4) {
        const float4 f = *(const float4*)&xr[i];
        acc += f.x + f.y + f.z + f.w;
    }
#pragma unroll
    for (int off = 16; off > 0; off >>= 1)
        acc += __shfl_xor(acc, off, 32);
    if (lane == 0)
        S[row] = acc;
}

// ---------------------------------------------------------------------------
// Main GEMM: acc = x_f16 * w_q_f16 (raw nibbles, exact), epilogue applies
// out = scale * (acc - zero * rowsum)
// ---------------------------------------------------------------------------
__global__ __launch_bounds__(256) void
q4_asym_wmma_gemm(const float* __restrict__ x,          // [M,K] fp32
                  const int*   __restrict__ wp,         // [K/2, N] packed nibbles
                  const float* __restrict__ scale,      // [N]
                  const float* __restrict__ zero,       // [N]
                  const float* __restrict__ S,          // [M] row sums of x
                  float*       __restrict__ out)        // [M,N] fp32
{
    // Double-buffered LDS. A row-major (rows of 32 f16); B column-major
    // (column n = 32 contiguous K halves): one packed int32 -> one ds_store_b32.
    __shared__ __align__(16) _Float16 As[2][BM * BK];   // 2 x 16 KB
    __shared__ __align__(16) _Float16 Bs[2][BN * BK];   // 2 x  8 KB

    const int tid  = threadIdx.x;
    const int lane = tid & 31;
    const int wave = tid >> 5;          // 0..7
    const int ln16 = lane & 15;
    const int hsel = lane >> 4;         // 0: lanes 0-15, 1: lanes 16-31

    const int mBase = blockIdx.y * BM;
    const int nBase = blockIdx.x * BN;

    // ---- A-fill: 4 passes of 64 rows; 8 f32 -> one v8h (ds_store_b128) ----
    const int aRow = tid >> 2;          // 0..63
    const int aCol = (tid & 3) * 8;     // 0,8,16,24

    // ---- B-fill: thread covers 4 adjacent columns at one packed row -------
    //      (global: int4 = 4 packed words, fully coalesced per wave)
    const int bN0 = (tid & 31) * 4;     // column group base
    const int bP0 = tid >> 5;           // packed row 0..7 (+8 on 2nd pass)

    // ---- wave -> 64x64 sub-tile (4 M-tiles x 4 N-tiles) --------------------
    const int mW = (wave & 3) * 64;
    const int nW = (wave >> 2) * 64;

    const v8f zero8 = {0.f, 0.f, 0.f, 0.f, 0.f, 0.f, 0.f, 0.f};
    v8f acc[4][4];
#pragma unroll
    for (int mt = 0; mt < 4; ++mt)
#pragma unroll
        for (int nt = 0; nt < 4; ++nt)
            acc[mt][nt] = zero8;

    // packed-f16 bias for the nibble->f16 exponent trick
    const v2h bias2 = {(_Float16)1024.f, (_Float16)1024.f};

    const int KT = KDIM / BK;           // 128 K-tiles

    for (int kt = 0; kt < KT; ++kt) {
        const int buf   = kt & 1;
        const int kBase = kt * BK;
        const int pRow  = kBase >> 1;   // packed-K row base (16 rows per tile)

        // -------- stage A tile: fp32 -> f16, row-major ----------------------
#pragma unroll
        for (int pp = 0; pp < 4; ++pp) {
            const int row = pp * 64 + aRow;
            const float4* px =
                (const float4*)&x[(size_t)(mBase + row) * KDIM + kBase + aCol];
            const float4 f0 = px[0];
            const float4 f1 = px[1];
            v8h h;
            h[0] = (_Float16)f0.x; h[1] = (_Float16)f0.y;
            h[2] = (_Float16)f0.z; h[3] = (_Float16)f0.w;
            h[4] = (_Float16)f1.x; h[5] = (_Float16)f1.y;
            h[6] = (_Float16)f1.z; h[7] = (_Float16)f1.w;
            *(v8h*)&As[buf][row * BK + aCol] = h;
        }

        // -------- stage B tile: nibbles -> exact f16 via exponent trick -----
        // pk = (v&0xF) | ((v&0xF0)<<12) | 0x64006400  => {1024+n0, 1024+n1}
        // then one v_pk_add_f16 (-1024) => {n0, n1}
#pragma unroll
        for (int j = 0; j < 2; ++j) {
            const int p = bP0 + j * 8;          // packed row within tile
            const int4 w4 =
                *(const int4*)&wp[(size_t)(pRow + p) * NDIM + nBase + bN0];
            const int wv[4] = {w4.x, w4.y, w4.z, w4.w};
#pragma unroll
            for (int c = 0; c < 4; ++c) {
                const unsigned v  = (unsigned)wv[c];
                const unsigned pk =
                    (v & 0xFu) | ((v & 0xF0u) << 12) | 0x64006400u;
                const v2h h = __builtin_bit_cast(v2h, pk) - bias2;
                *(v2h*)&Bs[buf][(bN0 + c) * BK + 2 * p] = h;
            }
        }

        __syncthreads();   // single barrier per K-tile (double buffered)

        // -------- prefetch next K tile (global_prefetch_b8) -----------------
        if (kt + 1 < KT) {
            __builtin_prefetch(
                &x[(size_t)(mBase + aRow) * KDIM + kBase + BK + aCol], 0, 1);
            __builtin_prefetch(
                &wp[(size_t)(pRow + 16 + bP0) * NDIM + nBase + bN0], 0, 1);
        }

        // -------- A fragments (ISA 16-bit A layout, 2x ds_load_b128 each) ---
        v16h a[4];
#pragma unroll
        for (int mt = 0; mt < 4; ++mt) {
            const int m  = mW + mt * 16 + ln16;
            const int kb = hsel ? 8 : 0;
            const uint4* pA = (const uint4*)&As[buf][m * BK + kb];
            ((uint4*)&a[mt])[0] = pA[0];   // K kb..kb+7
            ((uint4*)&a[mt])[1] = pA[2];   // K kb+16..kb+23
        }

        // -------- B fragments + 16x v_wmma_f32_16x16x32_f16 -----------------
#pragma unroll
        for (int nt = 0; nt < 4; ++nt) {
            const int n = nW + nt * 16 + ln16;
            const uint4* pB = (const uint4*)&Bs[buf][n * BK + hsel * 8];
            v16h b;
            ((uint4*)&b)[0] = pB[0];
            ((uint4*)&b)[1] = pB[2];
#pragma unroll
            for (int mt = 0; mt < 4; ++mt)
                acc[mt][nt] = __builtin_amdgcn_wmma_f32_16x16x32_f16(
                    false, a[mt], false, b,
                    (short)0, acc[mt][nt], false, false);
        }
    }

    // -------- epilogue: out = scale * (acc - zero * rowsum) ----------------
    // C layout: VGPR r, lanes 0-15 -> (M=r, N=lane); lanes 16-31 -> (M=r+8).
#pragma unroll
    for (int mt = 0; mt < 4; ++mt) {
        const int row0 = mBase + mW + mt * 16 + hsel * 8;
        float sv[8];
#pragma unroll
        for (int r = 0; r < 8; ++r)
            sv[r] = S[row0 + r];
#pragma unroll
        for (int nt = 0; nt < 4; ++nt) {
            const int col = nBase + nW + nt * 16 + ln16;
            const float sc = scale[col];
            const float zp = zero[col];
#pragma unroll
            for (int r = 0; r < 8; ++r)
                out[(size_t)(row0 + r) * NDIM + col] =
                    sc * (acc[mt][nt][r] - zp * sv[r]);
        }
    }
}

extern "C" void kernel_launch(void* const* d_in, const int* in_sizes, int n_in,
                              void* d_out, int out_size, void* d_ws, size_t ws_size,
                              hipStream_t stream) {
    const float* x     = (const float*)d_in[0];
    const int*   wp    = (const int*)d_in[1];
    const float* scale = (const float*)d_in[2];
    const float* zero  = (const float*)d_in[3];
    float*       out   = (float*)d_out;
    float*       S     = (float*)d_ws;          // 4096 floats = 16 KB

    rowsum_kernel<<<MDIM / 8, 256, 0, stream>>>(x, S);

    dim3 grid(NDIM / BN, MDIM / BM);            // 86 x 16 blocks
    dim3 block(256);                            // 8 wave32 waves
    q4_asym_wmma_gemm<<<grid, block, 0, stream>>>(x, wp, scale, zero, S, out);
}
